// TopKSAE_17875653886679
// MI455X (gfx1250) — compile-verified, matrix operand order checked by reading, not run on hardware
//
#include <hip/hip_runtime.h>
#include <hip/hip_bf16.h>

// ---------------------------------------------------------------------------
// Problem constants (from reference): B=4096, D=1024, F=16384, k=32
// ---------------------------------------------------------------------------
#define BSZ   4096
#define DDIM  1024
#define FDIM  16384
#define KSEL  32

// GEMM tiling
#define BM 128      // block tile M (batch)
#define BN 128      // block tile N (dict features)
#define KT 32       // K step (bf16 WMMA K=32)
#define LDK 40      // padded LDS row stride in bf16 units (80 B: 16B-aligned frags, spread banks)

#define A_PLANE (BM * LDK)            // 5120 halves = 10240 B
#define B_PLANE (BN * LDK)            // 5120 halves = 10240 B
#define BUF_HALVES (2 * A_PLANE + 2 * B_PLANE)   // Ahi|Alo|Bhi|Blo = 20480 halves
#define SMEM_HALVES (2 * BUF_HALVES)             // double buffered = 81920 B

typedef __bf16 bf16;
typedef __attribute__((ext_vector_type(16))) __bf16 v16bf;
typedef __attribute__((ext_vector_type(8)))  float  v8f;
typedef __attribute__((ext_vector_type(4)))  unsigned int u32x4;
typedef __attribute__((ext_vector_type(8)))  int i32x8;
typedef __attribute__((ext_vector_type(4)))  int i32x4;

#if defined(__has_builtin)
#if __has_builtin(__builtin_amdgcn_tensor_load_to_lds) && __has_builtin(__builtin_amdgcn_s_wait_tensorcnt)
#define HAVE_TDM 1
#endif
#endif

// Load a 16-element bf16 fragment (32 bytes) from LDS as two 16B chunks.
__device__ inline v16bf ld_frag(const bf16* p) {
  union { uint4 u[2]; v16bf v; } t;
  t.u[0] = *(const uint4*)(p);
  t.u[1] = *(const uint4*)(p + 8);
  return t.v;
}

// ---------------------------------------------------------------------------
// Kernel 0: one-shot split of (x - b_dec) and W_enc into bf16 hi/lo planes in
// d_ws. Removes all per-tile fp32->bf16 conversion from the GEMM hot loop and
// makes every k-step tile a plain 2D bf16 tile the TDM can DMA directly.
// ---------------------------------------------------------------------------
__global__ __launch_bounds__(256)
void sae_split_planes(const float* __restrict__ x,
                      const float* __restrict__ W_enc,
                      const float* __restrict__ b_dec,
                      bf16* __restrict__ xhi, bf16* __restrict__ xlo,
                      bf16* __restrict__ whi, bf16* __restrict__ wlo)
{
  const size_t XN = (size_t)BSZ * DDIM;
  const size_t WN = (size_t)FDIM * DDIM;
  const size_t stride = (size_t)gridDim.x * 256;
  for (size_t i = (size_t)blockIdx.x * 256 + threadIdx.x; i < XN + WN; i += stride) {
    if (i < XN) {
      float v = x[i] - b_dec[i & (DDIM - 1)];
      bf16 h = (bf16)v;
      xhi[i] = h;
      xlo[i] = (bf16)(v - (float)h);
    } else {
      size_t j = i - XN;
      float v = W_enc[j];
      bf16 h = (bf16)v;
      whi[j] = h;
      wlo[j] = (bf16)(v - (float)h);
    }
  }
}

#ifdef HAVE_TDM
// Issue one TDM 2D tile load: nrows x KT bf16 tile, row stride DDIM halves in
// memory, LDS rows padded 64B->80B via D# pad fields (interval=16 DW, pad=4 DW).
// This toolchain's builtin takes 6 args: (g0, g1, g2, g3, extra_g, cpol).
__device__ inline void tdm_load_tile(const bf16* gbase, int row0, int k0,
                                     unsigned nrows_tensor, unsigned nrows_tile,
                                     unsigned lds_byte_off) {
  unsigned long long ga =
      (unsigned long long)(const void*)gbase +
      (((unsigned long long)(unsigned)row0 * DDIM + (unsigned)k0) << 1);
  u32x4 g0;
  g0[0] = 1u;                                       // count=1 (valid user D#)
  g0[1] = lds_byte_off;                             // lds_addr
  g0[2] = (unsigned)(ga & 0xFFFFFFFFu);             // global_addr[31:0]
  g0[3] = (unsigned)((ga >> 32) & 0x01FFFFFFu)      // global_addr[56:32]
          | 0x80000000u;                            // type=2 ("image")
  i32x8 g1;
  g1[0] = (int)((1u << 16)                          // data_size = 2 bytes
              | (1u << 20)                          // pad_enable
              | (3u << 22)                          // pad_interval: 16 DWORDs
              | (3u << 25));                        // pad_amount:   4 DWORDs
  g1[1] = (int)((DDIM & 0xFFFFu) << 16);            // tensor_dim0 lo16 @ [63:48]
  g1[2] = (int)(((unsigned)DDIM >> 16) | ((nrows_tensor & 0xFFFFu) << 16));
  g1[3] = (int)((nrows_tensor >> 16) | ((unsigned)KT << 16));   // tile_dim0=32
  g1[4] = (int)nrows_tile;                          // tile_dim1, tile_dim2=0
  g1[5] = DDIM;                                     // tensor_dim0_stride lo32
  g1[6] = 0;
  g1[7] = 0;
  i32x4 gz4 = {0, 0, 0, 0};
  i32x8 gz8 = {0, 0, 0, 0, 0, 0, 0, 0};
  __builtin_amdgcn_tensor_load_to_lds(g0, g1, gz4, gz4, gz8, 0);
}
#endif

// ---------------------------------------------------------------------------
// Kernel 1: z_pre = (x - b_dec) @ W_enc^T + b_enc  via split-bf16 WMMA.
// Operands come pre-split from d_ws; tiles stream into double-buffered LDS via
// the Tensor Data Mover (TENSORcnt-tracked), 24 v_wmma per wave per k-step.
// ---------------------------------------------------------------------------
__global__ __launch_bounds__(256)
void sae_encode_wmma(const bf16* __restrict__ xhi, const bf16* __restrict__ xlo,
                     const bf16* __restrict__ whi, const bf16* __restrict__ wlo,
                     const float* __restrict__ b_enc,
                     float* __restrict__ z_pre)
{
  __shared__ __align__(16) bf16 smem[SMEM_HALVES];   // 81920 B of the 320 KB WGP LDS

  const int t    = threadIdx.x;
  const int lane = t & 31;
  const int wid  = t >> 5;
  const int wm   = wid & 3;          // wave grid: 4 (M) x 2 (N)
  const int wn   = wid >> 2;
  const int m0   = blockIdx.y * BM;
  const int n0   = blockIdx.x * BN;

  const int lrow = lane & 15;                 // row within 16x16 frag
  const int kofs = (lane >> 4) << 4;          // lanes 16..31 hold K=16..31

  // 2 (M) x 4 (N) accumulator fragments, seeded with b_enc broadcast.
  v8f acc[2][4];
#pragma unroll
  for (int ni = 0; ni < 4; ++ni) {
    float be = b_enc[n0 + wn * 64 + ni * 16 + lrow];
#pragma unroll
    for (int mi = 0; mi < 2; ++mi)
#pragma unroll
      for (int v = 0; v < 8; ++v) acc[mi][ni][v] = be;
  }

  const unsigned smem_base = (unsigned)(size_t)(&smem[0]);  // LDS byte offset

#ifdef HAVE_TDM
  // Prologue: DMA tile 0 into buffer 0 (wave 0 issues; EXEC ignored by TDM).
  if (wid == 0) {
    tdm_load_tile(xhi, m0, 0, BSZ,  BM, smem_base + 0 * (BUF_HALVES * 2));
    tdm_load_tile(xlo, m0, 0, BSZ,  BM, smem_base + 0 + A_PLANE * 2);
    tdm_load_tile(whi, n0, 0, FDIM, BN, smem_base + 0 + 2 * A_PLANE * 2);
    tdm_load_tile(wlo, n0, 0, FDIM, BN, smem_base + 0 + 2 * A_PLANE * 2 + B_PLANE * 2);
  }
#endif

  for (int step = 0; step < DDIM / KT; ++step) {
    const int cur = step & 1;
    const unsigned curh = (unsigned)cur * BUF_HALVES;

#ifdef HAVE_TDM
    if (wid == 0) {
      if (step + 1 < DDIM / KT) {
        const unsigned nb = smem_base + (unsigned)((step + 1) & 1) * (BUF_HALVES * 2);
        const int k1 = (step + 1) * KT;
        tdm_load_tile(xhi, m0, k1, BSZ,  BM, nb);
        tdm_load_tile(xlo, m0, k1, BSZ,  BM, nb + A_PLANE * 2);
        tdm_load_tile(whi, n0, k1, FDIM, BN, nb + 2 * A_PLANE * 2);
        tdm_load_tile(wlo, n0, k1, FDIM, BN, nb + 2 * A_PLANE * 2 + B_PLANE * 2);
        __builtin_amdgcn_s_wait_tensorcnt(4);   // tile `step` complete, 4 in flight
      } else {
        __builtin_amdgcn_s_wait_tensorcnt(0);
      }
    }
    __syncthreads();   // tile `step` visible in LDS to all waves
#else
    // Fallback: cooperative 16B-chunk copies from the bf16 planes into LDS.
    {
      const int k0 = step * KT;
      for (int c = t; c < 4 * BM * (KT / 8); c += 256) {
        const int plane = c >> 9;            // 0:Ahi 1:Alo 2:Bhi 3:Blo (512 chunks ea)
        const int r     = (c >> 2) & (BM - 1);
        const int ch    = c & 3;
        const bf16* gb  = plane == 0 ? xhi : plane == 1 ? xlo : plane == 2 ? whi : wlo;
        const int row0  = plane < 2 ? m0 : n0;
        const unsigned dsth = curh + (unsigned)plane * A_PLANE + r * LDK + ch * 8;
        *(uint4*)(&smem[dsth]) =
            *(const uint4*)(gb + (size_t)(row0 + r) * DDIM + k0 + ch * 8);
      }
      __syncthreads();
    }
#endif

    v16bf ah[2], al[2], bh[4], bl[4];
#pragma unroll
    for (int mi = 0; mi < 2; ++mi) {
      const int r = wm * 32 + mi * 16 + lrow;
      ah[mi] = ld_frag(&smem[curh + 0 * A_PLANE + r * LDK + kofs]);
      al[mi] = ld_frag(&smem[curh + 1 * A_PLANE + r * LDK + kofs]);
    }
#pragma unroll
    for (int ni = 0; ni < 4; ++ni) {
      const int r = wn * 64 + ni * 16 + lrow;
      bh[ni] = ld_frag(&smem[curh + 2 * A_PLANE + 0 * B_PLANE + r * LDK + kofs]);
      bl[ni] = ld_frag(&smem[curh + 2 * A_PLANE + 1 * B_PLANE + r * LDK + kofs]);
    }

#pragma unroll
    for (int mi = 0; mi < 2; ++mi)
#pragma unroll
      for (int ni = 0; ni < 4; ++ni) {
        acc[mi][ni] = __builtin_amdgcn_wmma_f32_16x16x32_bf16(
            false, ah[mi], false, bh[ni], (short)0, acc[mi][ni], false, false);
        acc[mi][ni] = __builtin_amdgcn_wmma_f32_16x16x32_bf16(
            false, ah[mi], false, bl[ni], (short)0, acc[mi][ni], false, false);
        acc[mi][ni] = __builtin_amdgcn_wmma_f32_16x16x32_bf16(
            false, al[mi], false, bh[ni], (short)0, acc[mi][ni], false, false);
      }
    __syncthreads();   // compute done before buffer `cur` is overwritten
  }

  // Store: C layout = lanes 0-15: (M=v, N=lane); lanes 16-31: (M=v+8, N=lane-16)
#pragma unroll
  for (int mi = 0; mi < 2; ++mi)
#pragma unroll
    for (int ni = 0; ni < 4; ++ni) {
      const int col   = n0 + wn * 64 + ni * 16 + lrow;
      const int rbase = m0 + wm * 32 + mi * 16 + ((lane >> 4) << 3);
#pragma unroll
      for (int v = 0; v < 8; ++v)
        z_pre[(size_t)(rbase + v) * FDIM + col] = acc[mi][ni][v];
    }
}

// ---------------------------------------------------------------------------
// Kernel 2: per-row exact TopK(k=32) via 4-pass radix select on monotonic
// float keys, then z scatter + sparse decode x_hat = z @ W_dec^T + b_dec.
// One block per batch row; row staged in LDS (64 KB of the 320 KB/WGP).
// ---------------------------------------------------------------------------
__device__ inline unsigned f2key(float f) {
  unsigned u = __float_as_uint(f);
  return u ^ (unsigned)((((int)u) >> 31) | 0x80000000);
}

__global__ __launch_bounds__(256)
void sae_topk_decode(const float* __restrict__ z_pre,
                     const float* __restrict__ W_dec,
                     const float* __restrict__ b_dec,
                     float* __restrict__ z,
                     float* __restrict__ x_hat)
{
  __shared__ float    vals[FDIM];        // 64 KB row stage
  __shared__ unsigned hist[256];
  __shared__ unsigned sPrefix, sRemain;  // radix-select state
  __shared__ unsigned nsel, eqTaken;
  __shared__ int      selIdx[KSEL];
  __shared__ float    selVal[KSEL];

  const int b = blockIdx.x;
  const int t = threadIdx.x;
  const float* row = z_pre + (size_t)b * FDIM;
  float* zrow = z + (size_t)b * FDIM;

  for (int j = t; j < FDIM; j += 256) {
    float v = row[j];
    vals[j] = v;
    zrow[j] = 0.0f;
  }
  if (t == 0) { sPrefix = 0u; sRemain = KSEL; nsel = 0u; eqTaken = 0u; }
  if (t < KSEL) { selIdx[t] = 0; selVal[t] = 0.0f; }
  __syncthreads();

  for (int pass = 0; pass < 4; ++pass) {
    const int shift = 24 - 8 * pass;
    hist[t & 255] = 0u;
    __syncthreads();
    const unsigned pfx = sPrefix;
    for (int j = t; j < FDIM; j += 256) {
      unsigned key = f2key(vals[j]);
      if (pass == 0 || (key >> (shift + 8)) == pfx)
        atomicAdd(&hist[(key >> shift) & 255u], 1u);
    }
    __syncthreads();
    if (t == 0) {
      unsigned rem = sRemain, cum = 0u;
      int bin = 255;
      for (; bin > 0; --bin) {
        unsigned c = hist[bin];
        if (cum + c >= rem) break;
        cum += c;
      }
      sPrefix = (sPrefix << 8) | (unsigned)bin;
      sRemain = rem - cum;
    }
    __syncthreads();
  }

  const unsigned T  = sPrefix;
  const unsigned eq = sRemain;

  for (int j = t; j < FDIM; j += 256) {
    unsigned key = f2key(vals[j]);
    if (key > T) {
      unsigned p = atomicAdd(&nsel, 1u);
      if (p < KSEL) { selIdx[p] = j; selVal[p] = vals[j]; }
    } else if (key == T) {
      unsigned e = atomicAdd(&eqTaken, 1u);
      if (e < eq) {
        unsigned p = atomicAdd(&nsel, 1u);
        if (p < KSEL) { selIdx[p] = j; selVal[p] = vals[j]; }
      }
    }
  }
  __syncthreads();

  const int n = (int)(nsel < (unsigned)KSEL ? nsel : (unsigned)KSEL);

  if (t < n) {
    float v = selVal[t];
    v = v > 0.0f ? v : 0.0f;
    selVal[t] = v;
    zrow[selIdx[t]] = v;
  }
  __syncthreads();

  for (int d = t; d < DDIM; d += 256) {
    float acc = b_dec[d];
    const float* wrow = W_dec + (size_t)d * FDIM;
#pragma unroll 8
    for (int i = 0; i < KSEL; ++i)
      acc += selVal[i] * wrow[selIdx[i]];
    x_hat[(size_t)b * DDIM + d] = acc;
  }
}

// ---------------------------------------------------------------------------
// Launcher. Inputs: x, W_enc, b_enc, W_dec, b_dec, k.
// d_out = [x_hat (B*D) | z (B*F) | z_pre (B*F)] fp32, concatenated flat.
// d_ws  = [xhi | xlo | whi | wlo] bf16 planes (~84 MB).
// ---------------------------------------------------------------------------
extern "C" void kernel_launch(void* const* d_in, const int* in_sizes, int n_in,
                              void* d_out, int out_size, void* d_ws, size_t ws_size,
                              hipStream_t stream) {
  const float* x     = (const float*)d_in[0];
  const float* W_enc = (const float*)d_in[1];
  const float* b_enc = (const float*)d_in[2];
  const float* W_dec = (const float*)d_in[3];
  const float* b_dec = (const float*)d_in[4];
  (void)in_sizes; (void)n_in; (void)ws_size; (void)out_size;

  float* x_hat = (float*)d_out;
  float* z     = x_hat + (size_t)BSZ * DDIM;
  float* z_pre = z + (size_t)BSZ * FDIM;

  const size_t XN = (size_t)BSZ * DDIM;
  const size_t WN = (size_t)FDIM * DDIM;
  bf16* xhi = (bf16*)d_ws;
  bf16* xlo = xhi + XN;
  bf16* whi = xlo + XN;
  bf16* wlo = whi + WN;

  sae_split_planes<<<2048, 256, 0, stream>>>(x, W_enc, b_dec, xhi, xlo, whi, wlo);

  dim3 g1(FDIM / BN, BSZ / BM);   // 128 x 32 blocks
  sae_encode_wmma<<<g1, 256, 0, stream>>>(xhi, xlo, whi, wlo, b_enc, z_pre);

  sae_topk_decode<<<BSZ, 256, 0, stream>>>(z_pre, W_dec, b_dec, z, x_hat);
}
